// SparselyGatedMoELayer_35699768164691
// MI455X (gfx1250) — compile-verified, matrix operand order checked by reading, not run on hardware
//
#include <hip/hip_runtime.h>
#include <math.h>

#define B_TOK 8192
#define D_DIM 1024
#define E_NUM 8
#define H_DIM 4096
#define K_TOP 2

#define BM 128
#define BN 256
#define BK 32
#define LDT 40   // padded LDS row stride in halves (32 data + 8 pad) -> conflict-free b128

typedef __attribute__((ext_vector_type(16))) __bf16 v16bf;
typedef __attribute__((ext_vector_type(8)))  __bf16 v8bf;
typedef __attribute__((ext_vector_type(8)))  float  v8f;

__device__ __forceinline__ unsigned short f2bf(float f) {
    unsigned int u = __float_as_uint(f);
    unsigned int r = u + 0x7FFFu + ((u >> 16) & 1u);   // round-to-nearest-even
    return (unsigned short)(r >> 16);
}

__device__ __forceinline__ v8f wmma_bf16(v16bf a, v16bf b, v8f c) {
    return __builtin_amdgcn_wmma_f32_16x16x32_bf16(
        /*neg_a=*/false, a, /*neg_b=*/false, b,
        /*c_mod=*/(short)0, c, /*reuse_a=*/false, /*reuse_b=*/false);
}

// Load one 16x32 bf16 fragment from LDS (row-major, LDT-half row stride).
// ISA layout (CDNA5 ISA): lanes 0-15 hold K={0..7,16..23}; lanes 16-31 hold K={8..15,24..31}.
__device__ __forceinline__ v16bf ldfrag(const unsigned short* p) {
    v8bf lo = *(const v8bf*)(const void*)p;          // K kb..kb+7
    v8bf hi = *(const v8bf*)(const void*)(p + 16);   // K 16+kb..16+kb+7
    v16bf r;
#pragma unroll
    for (int i = 0; i < 8; ++i) { r[i] = lo[i]; r[8 + i] = hi[i]; }
    return r;
}

// ---------------- small kernels ----------------

__global__ void moe_zero_counts(int* counts) {
    if (threadIdx.x < E_NUM) counts[threadIdx.x] = 0;
}

__global__ void moe_cvt_x(const float4* __restrict__ x, ushort4* __restrict__ xb) {
    int i = blockIdx.x * blockDim.x + threadIdx.x;   // over B*D/4
    float4 v = x[i];
    ushort4 o;
    o.x = f2bf(v.x); o.y = f2bf(v.y); o.z = f2bf(v.z); o.w = f2bf(v.w);
    xb[i] = o;
}

// transpose-convert: per-expert slice src[R][C] fp32 -> dst[C][R] bf16
__global__ void moe_cvt_t(const float* __restrict__ src, unsigned short* __restrict__ dst,
                          int R, int C) {
    __shared__ float tile[32][33];
    size_t slice = (size_t)blockIdx.z * (size_t)R * (size_t)C;
    int tx = threadIdx.x, ty = threadIdx.y;
    int c0 = blockIdx.x * 32, r0 = blockIdx.y * 32;
    const float* s = src + slice;
    unsigned short* d = dst + slice;
#pragma unroll
    for (int i = 0; i < 32; i += 8)
        tile[ty + i][tx] = s[(size_t)(r0 + ty + i) * C + c0 + tx];
    __syncthreads();
#pragma unroll
    for (int i = 0; i < 32; i += 8)
        d[(size_t)(c0 + ty + i) * R + r0 + tx] = f2bf(tile[tx][ty + i]);
}

// noisy top-2 gate; wave-per-token
__global__ void moe_gate(const float* __restrict__ xg, const float* __restrict__ noise,
                         const float* __restrict__ gw, const float* __restrict__ nw,
                         float* __restrict__ g_sel, int* __restrict__ list,
                         int* __restrict__ counts) {
    int t = blockIdx.x * 8 + (threadIdx.x >> 5);
    int lane = threadIdx.x & 31;
    float ag[E_NUM], an[E_NUM];
#pragma unroll
    for (int e = 0; e < E_NUM; ++e) { ag[e] = 0.f; an[e] = 0.f; }
    const float* xr = xg + (size_t)t * D_DIM;
    for (int d = lane; d < D_DIM; d += 32) {
        float xv = xr[d];
#pragma unroll
        for (int e = 0; e < E_NUM; ++e) {
            ag[e] = fmaf(xv, gw[e * D_DIM + d], ag[e]);
            an[e] = fmaf(xv, nw[e * D_DIM + d], an[e]);
        }
    }
    for (int off = 16; off > 0; off >>= 1) {
#pragma unroll
        for (int e = 0; e < E_NUM; ++e) {
            ag[e] += __shfl_xor(ag[e], off, 32);
            an[e] += __shfl_xor(an[e], off, 32);
        }
    }
    if (lane == 0) {
        float lg[E_NUM];
#pragma unroll
        for (int e = 0; e < E_NUM; ++e) {
            float x = an[e];
            float sp = (x > 20.f) ? x : log1pf(expf(x));     // softplus
            lg[e] = ag[e] + noise[t * E_NUM + e] * sp;
        }
        int i0 = 0;
#pragma unroll
        for (int e = 1; e < E_NUM; ++e) if (lg[e] > lg[i0]) i0 = e;
        int i1 = (i0 == 0) ? 1 : 0;
#pragma unroll
        for (int e = 0; e < E_NUM; ++e) if (e != i0 && lg[e] > lg[i1]) i1 = e;
        float d1 = expf(lg[i1] - lg[i0]);
        float s = 1.f + d1;
        float g0 = 1.f / s, g1 = d1 / s;
        int s0 = atomicAdd(&counts[i0], 1); list[i0 * B_TOK + s0] = t * 2;
        int s1 = atomicAdd(&counts[i1], 1); list[i1 * B_TOK + s1] = t * 2 + 1;
        g_sel[t * 2] = g0; g_sel[t * 2 + 1] = g1;
    }
}

// ---------------- main expert GEMM (routed, WMMA bf16) ----------------
// MODE 1: h[r] = relu(xb[token(r)] @ w1t[e]^T + b1[e])   (K=D, N=H)  -> bf16
// MODE 2: y[r] =       h[r]        @ w2t[e]^T + b2[e]    (K=H, N=D)  -> f32
// Block tile 128x256, 8 waves as 2(M)x4(N), each wave a 64x64 tile:
// 16 WMMAs per 16 ds_load_b128 per K-step.
template <int MODE>
__global__ __launch_bounds__(256) void moe_gemm(
    const unsigned short* __restrict__ A,   // bf16 activations (xb or h)
    const unsigned short* __restrict__ W,   // bf16 weights, N-major: [E][N][K]
    const float* __restrict__ bias,         // [E][N]
    const int* __restrict__ list, const int* __restrict__ counts,
    unsigned short* __restrict__ hout,      // MODE 1
    float* __restrict__ yout)               // MODE 2
{
    constexpr int Kdim = (MODE == 1) ? D_DIM : H_DIM;
    constexpr int Ndim = (MODE == 1) ? H_DIM : D_DIM;
    const int e = blockIdx.z;
    const int cnt = counts[e];
    const int m0 = blockIdx.y * BM;
    if (m0 >= cnt) return;
    const int n0 = blockIdx.x * BN;

    __shared__ __align__(16) unsigned short sA[2][BM * LDT];
    __shared__ __align__(16) unsigned short sB[2][BN * LDT];
    __shared__ int rows[BM];

    const int tid = threadIdx.x;
    if (tid < BM) {
        int slot = m0 + tid;
        rows[tid] = (slot < cnt) ? list[e * B_TOK + slot] : list[e * B_TOK];
    }
    __syncthreads();

    // global->LDS staging: A thread covers rows (lr, lr+64); B rows (lr + 64*g, g<4)
    const int lr = tid >> 2;
    const int lc = (tid & 3) * 8;
    long aR0 = (MODE == 1) ? (rows[lr] >> 1) : rows[lr];
    long aR1 = (MODE == 1) ? (rows[lr + 64] >> 1) : rows[lr + 64];
    const unsigned short* aPtr0 = A + aR0 * (long)Kdim + lc;
    const unsigned short* aPtr1 = A + aR1 * (long)Kdim + lc;
    const unsigned short* bPtr = W + (long)e * Kdim * Ndim + (long)(n0 + lr) * Kdim + lc;

    const int wid = tid >> 5, lane = tid & 31;
    const int wm = (wid >> 2) * 64;      // wave M origin (2 waves in M)
    const int wn = (wid & 3) * 64;       // wave N origin (4 waves in N)
    const int l15 = lane & 15;
    const int hi = lane >> 4;
    const int kb = hi * 8;

    v8f acc[4][4];
    v8f vz = {0.f, 0.f, 0.f, 0.f, 0.f, 0.f, 0.f, 0.f};
#pragma unroll
    for (int i = 0; i < 4; ++i)
#pragma unroll
        for (int j = 0; j < 4; ++j) acc[i][j] = vz;

    constexpr int NK = Kdim / BK;
    uint4 ra0 = *(const uint4*)(const void*)aPtr0;
    uint4 ra1 = *(const uint4*)(const void*)aPtr1;
    uint4 rb[4];
#pragma unroll
    for (int g = 0; g < 4; ++g)
        rb[g] = *(const uint4*)(const void*)(bPtr + 64L * g * Kdim);
    *(uint4*)(void*)&sA[0][lr * LDT + lc] = ra0;
    *(uint4*)(void*)&sA[0][(lr + 64) * LDT + lc] = ra1;
#pragma unroll
    for (int g = 0; g < 4; ++g)
        *(uint4*)(void*)&sB[0][(lr + 64 * g) * LDT + lc] = rb[g];
    __syncthreads();

    for (int kt = 0; kt < NK; ++kt) {
        const int cur = kt & 1;
        if (kt + 1 < NK) {
            const int ko = (kt + 1) * BK;
            ra0 = *(const uint4*)(const void*)(aPtr0 + ko);
            ra1 = *(const uint4*)(const void*)(aPtr1 + ko);
#pragma unroll
            for (int g = 0; g < 4; ++g)
                rb[g] = *(const uint4*)(const void*)(bPtr + ko + 64L * g * Kdim);
        }
        v16bf af[4], bfr[4];
#pragma unroll
        for (int i = 0; i < 4; ++i)
            af[i] = ldfrag(&sA[cur][(wm + i * 16 + l15) * LDT + kb]);
#pragma unroll
        for (int j = 0; j < 4; ++j)
            bfr[j] = ldfrag(&sB[cur][(wn + j * 16 + l15) * LDT + kb]);
#pragma unroll
        for (int i = 0; i < 4; ++i)
#pragma unroll
            for (int j = 0; j < 4; ++j)
                acc[i][j] = wmma_bf16(af[i], bfr[j], acc[i][j]);
        if (kt + 1 < NK) {
            __syncthreads();
            const int nx = cur ^ 1;
            *(uint4*)(void*)&sA[nx][lr * LDT + lc] = ra0;
            *(uint4*)(void*)&sA[nx][(lr + 64) * LDT + lc] = ra1;
#pragma unroll
            for (int g = 0; g < 4; ++g)
                *(uint4*)(void*)&sB[nx][(lr + 64 * g) * LDT + lc] = rb[g];
            __syncthreads();
        }
    }

    // epilogue: C/D layout -> VGPR j: lanes0-15 M=j, lanes16-31 M=j+8; N = lane&15
#pragma unroll
    for (int j = 0; j < 4; ++j) {
        const int ng = n0 + wn + j * 16 + l15;
        const float bv = bias[e * Ndim + ng];
#pragma unroll
        for (int i = 0; i < 4; ++i) {
#pragma unroll
            for (int rj = 0; rj < 8; ++rj) {
                const int ml = wm + i * 16 + rj + 8 * hi;
                if (m0 + ml < cnt) {
                    const int r = rows[ml];
                    float v = acc[i][j][rj] + bv;
                    if (MODE == 1) {
                        v = v > 0.f ? v : 0.f;
                        hout[(size_t)r * H_DIM + ng] = f2bf(v);
                    } else {
                        yout[(size_t)r * D_DIM + ng] = v;
                    }
                }
            }
        }
    }
}

// out[t] = g0 * y[2t] + g1 * y[2t+1]  (deterministic combine, float4)
__global__ void moe_combine(const float4* __restrict__ y, const float* __restrict__ g_sel,
                            float4* __restrict__ out) {
    int i = blockIdx.x * blockDim.x + threadIdx.x;   // over B*D/4
    int t = (i * 4) / D_DIM;
    int q = i - t * (D_DIM / 4);
    float g0 = g_sel[2 * t], g1 = g_sel[2 * t + 1];
    float4 a = y[(size_t)(2 * t) * (D_DIM / 4) + q];
    float4 b = y[(size_t)(2 * t + 1) * (D_DIM / 4) + q];
    float4 o;
    o.x = g0 * a.x + g1 * b.x;
    o.y = g0 * a.y + g1 * b.y;
    o.z = g0 * a.z + g1 * b.z;
    o.w = g0 * a.w + g1 * b.w;
    out[i] = o;
}

extern "C" void kernel_launch(void* const* d_in, const int* in_sizes, int n_in,
                              void* d_out, int out_size, void* d_ws, size_t ws_size,
                              hipStream_t stream) {
    (void)in_sizes; (void)n_in; (void)out_size; (void)ws_size;
    const float* x_gate    = (const float*)d_in[0];
    const float* x_experts = (const float*)d_in[1];
    const float* noise     = (const float*)d_in[2];
    const float* gate_w    = (const float*)d_in[3];
    const float* noise_w   = (const float*)d_in[4];
    const float* w1        = (const float*)d_in[5];
    const float* b1        = (const float*)d_in[6];
    const float* w2        = (const float*)d_in[7];
    const float* b2        = (const float*)d_in[8];
    float* out = (float*)d_out;

    char* ws = (char*)d_ws;
    size_t off = 0;
    int*            counts = (int*)(ws + off);            off += 256;
    float*          g_sel  = (float*)(ws + off);          off += (size_t)B_TOK * K_TOP * 4;
    int*            list   = (int*)(ws + off);            off += (size_t)E_NUM * B_TOK * 4;
    unsigned short* xb     = (unsigned short*)(ws + off); off += (size_t)B_TOK * D_DIM * 2;
    unsigned short* w1t    = (unsigned short*)(ws + off); off += (size_t)E_NUM * H_DIM * D_DIM * 2;
    unsigned short* w2t    = (unsigned short*)(ws + off); off += (size_t)E_NUM * H_DIM * D_DIM * 2;
    unsigned short* hbuf   = (unsigned short*)(ws + off); off += (size_t)B_TOK * K_TOP * H_DIM * 2;
    float*          ybuf   = (float*)(ws + off);          off += (size_t)B_TOK * K_TOP * D_DIM * 4;

    moe_zero_counts<<<1, 32, 0, stream>>>(counts);

    moe_cvt_x<<<(B_TOK * D_DIM / 4) / 256, 256, 0, stream>>>(
        (const float4*)x_experts, (ushort4*)xb);

    // w1 [E][D][H] -> w1t [E][H][D]
    moe_cvt_t<<<dim3(H_DIM / 32, D_DIM / 32, E_NUM), dim3(32, 8), 0, stream>>>(
        w1, w1t, D_DIM, H_DIM);
    // w2 [E][H][D] -> w2t [E][D][H]
    moe_cvt_t<<<dim3(D_DIM / 32, H_DIM / 32, E_NUM), dim3(32, 8), 0, stream>>>(
        w2, w2t, H_DIM, D_DIM);

    moe_gate<<<B_TOK / 8, 256, 0, stream>>>(x_gate, noise, gate_w, noise_w,
                                            g_sel, list, counts);

    moe_gemm<1><<<dim3(H_DIM / BN, B_TOK / BM, E_NUM), 256, 0, stream>>>(
        xb, w1t, b1, list, counts, hbuf, nullptr);

    moe_gemm<2><<<dim3(D_DIM / BN, B_TOK / BM, E_NUM), 256, 0, stream>>>(
        hbuf, w2t, b2, list, counts, nullptr, ybuf);

    moe_combine<<<(B_TOK * D_DIM / 4) / 256, 256, 0, stream>>>(
        (const float4*)ybuf, g_sel, (float4*)out);
}
